// EdgeConv_62354335203514
// MI455X (gfx1250) — matrix operand order repaired and checked by reading, not compiled, not run on hardware
//
#include <hip/hip_runtime.h>
#include <hip/hip_bf16.h>

typedef __attribute__((ext_vector_type(16))) _Float16 v16h;
typedef __attribute__((ext_vector_type(8)))  _Float16 v8h;
typedef __attribute__((ext_vector_type(8)))  float    v8f;

#define WMMA_F16(a, b, c) \
    __builtin_amdgcn_wmma_f32_16x16x32_f16(false, (a), false, (b), (short)0, (c), false, false)

static constexpr int B_  = 8;
static constexpr int C_  = 64;     // input channels (K dim = 64 => 2 WMMA k-chunks)
static constexpr int N_  = 4096;
static constexpr int K_  = 20;     // neighbors
static constexpr int CO_ = 64;     // output channels

// ---------------------------------------------------------------------------
// 1) Transpose x (B,C,N)->(B,N,C) as f16, plus squared norms xx[b,n].
// ---------------------------------------------------------------------------
__global__ void __launch_bounds__(256) prep_x(const float* __restrict__ x,
                                              _Float16* __restrict__ xh,
                                              float* __restrict__ xx) {
    int gid = blockIdx.x * 256 + threadIdx.x;       // b*N + n
    int b = gid >> 12;
    int n = gid & (N_ - 1);
    const float* xp = x + (size_t)b * C_ * N_ + n;
    _Float16* hp = xh + (size_t)gid * C_;
    float s = 0.f;
#pragma unroll
    for (int c = 0; c < C_; ++c) {
        float v = xp[(size_t)c * N_];
        s += v * v;
        hp[c] = (_Float16)v;
    }
    xx[gid] = s;
}

// ---------------------------------------------------------------------------
// 2) Pre-swizzle W into WMMA B-fragments for W1 and (W2 - W1).
//    Fragment id = (kchunk*4 + ntile)*32 + lane, 16 halves each.
//    B layout: lane (n = lane&15, h = lane>>4), element e -> k = h*8 + e (e<8), 16 + h*8 + (e-8) (e>=8).
// ---------------------------------------------------------------------------
__global__ void __launch_bounds__(256) prep_w(const float* __restrict__ W,
                                              _Float16* __restrict__ w1f,
                                              _Float16* __restrict__ wdf) {
    int t = threadIdx.x;            // 256 = 2 kchunks * 4 ntiles * 32 lanes
    int lane  = t & 31;
    int ntile = (t >> 5) & 3;
    int q     = t >> 7;
    int n = ntile * 16 + (lane & 15);
    int h = lane >> 4;
    _Float16* o1 = w1f + (size_t)t * 16;
    _Float16* od = wdf + (size_t)t * 16;
#pragma unroll
    for (int e = 0; e < 16; ++e) {
        int k = q * 32 + (e < 8 ? h * 8 + e : 16 + h * 8 + (e - 8));   // c index
        float w1 = W[n * 128 + k];
        float w2 = W[n * 128 + 64 + k];
        o1[e] = (_Float16)w1;
        od[e] = (_Float16)(w2 - w1);
    }
}

// ---------------------------------------------------------------------------
// 3) z = x . W1^T  and  y = x . (W2-W1)^T   (both (B*N,64) f32), via WMMA.
//    One wave per 16-row tile, 4 column tiles, K = 64 (2 chunks of 32).
// ---------------------------------------------------------------------------
__global__ void __launch_bounds__(32) zy_gemm(const _Float16* __restrict__ xh,
                                              const _Float16* __restrict__ w1f,
                                              const _Float16* __restrict__ wdf,
                                              float* __restrict__ z,
                                              float* __restrict__ y) {
    int bx   = blockIdx.x;
    int b    = bx >> 8;                 // 256 row-tiles per batch
    int row0 = (bx & 255) << 4;
    int lane = threadIdx.x;
    int m = lane & 15, h = lane >> 4;

    const _Float16* arow = xh + ((size_t)(b * N_ + row0 + m)) * C_;
    v16h a[2];
#pragma unroll
    for (int q = 0; q < 2; ++q) {
        v8h lo = *(const v8h*)(arow + q * 32 + h * 8);
        v8h hi = *(const v8h*)(arow + q * 32 + 16 + h * 8);
#pragma unroll
        for (int e = 0; e < 8; ++e) { a[q][e] = lo[e]; a[q][8 + e] = hi[e]; }
    }

#pragma unroll
    for (int nt = 0; nt < 4; ++nt) {
        v16h b10 = *(const v16h*)(w1f + (size_t)((0 * 4 + nt) * 32 + lane) * 16);
        v16h b11 = *(const v16h*)(w1f + (size_t)((1 * 4 + nt) * 32 + lane) * 16);
        v16h bd0 = *(const v16h*)(wdf + (size_t)((0 * 4 + nt) * 32 + lane) * 16);
        v16h bd1 = *(const v16h*)(wdf + (size_t)((1 * 4 + nt) * 32 + lane) * 16);
        v8f accz = {}; accz = WMMA_F16(a[0], b10, accz); accz = WMMA_F16(a[1], b11, accz);
        v8f accy = {}; accy = WMMA_F16(a[0], bd0, accy); accy = WMMA_F16(a[1], bd1, accy);
        // C layout: VGPR r -> (M = r + 8*(lane>>4), N = lane&15)
        int col   = nt * 16 + (lane & 15);
        int rbase = row0 + 8 * h;
#pragma unroll
        for (int r = 0; r < 8; ++r) {
            size_t off = ((size_t)(b * N_ + rbase + r)) * CO_ + col;
            z[off] = accz[r];
            y[off] = accy[r];
        }
    }
}

// ---------------------------------------------------------------------------
// 4) KNN: one wave per 32 rows. Gram tiles via WMMA, dist through LDS,
//    per-lane top-20 (smallest distance) selection. Self is included
//    (matches reference top_k over -dist).
// ---------------------------------------------------------------------------
__global__ void __launch_bounds__(32) knn_kernel(const _Float16* __restrict__ xh,
                                                 const float* __restrict__ xx,
                                                 int* __restrict__ idx) {
    __shared__ float sxx[N_];
    __shared__ float stile[2][16][16];

    int bx   = blockIdx.x;
    int b    = bx >> 7;                 // 128 groups of 32 rows per batch
    int row0 = (bx & 127) << 5;
    int lane = threadIdx.x;

    const _Float16* xb  = xh + (size_t)b * N_ * C_;
    const float*    xxb = xx + (size_t)b * N_;
    for (int i = lane; i < N_; i += 32) sxx[i] = xxb[i];
    __syncthreads();

    int m = lane & 15, h = lane >> 4;
    v16h a0[2], a1[2];                  // rows row0+m and row0+16+m
    const _Float16* ar0 = xb + (size_t)(row0 + m) * C_;
    const _Float16* ar1 = xb + (size_t)(row0 + 16 + m) * C_;
#pragma unroll
    for (int q = 0; q < 2; ++q) {
        v8h lo0 = *(const v8h*)(ar0 + q * 32 + h * 8);
        v8h hi0 = *(const v8h*)(ar0 + q * 32 + 16 + h * 8);
        v8h lo1 = *(const v8h*)(ar1 + q * 32 + h * 8);
        v8h hi1 = *(const v8h*)(ar1 + q * 32 + 16 + h * 8);
#pragma unroll
        for (int e = 0; e < 8; ++e) {
            a0[q][e] = lo0[e]; a0[q][8 + e] = hi0[e];
            a1[q][e] = lo1[e]; a1[q][8 + e] = hi1[e];
        }
    }

    float bd[K_]; int bi[K_];
#pragma unroll
    for (int i = 0; i < K_; ++i) { bd[i] = 3.0e38f; bi[i] = 0; }
    float worst = 3.0e38f;
    int wi = 0;

    float xr  = sxx[row0 + lane];
    int myt = lane >> 4, myr = lane & 15;

    for (int col0 = 0; col0 < N_; col0 += 16) {
        const _Float16* brow = xb + (size_t)(col0 + m) * C_;
        v8h bl0 = *(const v8h*)(brow + h * 8), bh0 = *(const v8h*)(brow + 16 + h * 8);
        v8h bl1 = *(const v8h*)(brow + 32 + h * 8), bh1 = *(const v8h*)(brow + 48 + h * 8);
        v16h bf0, bf1;
        for (int e = 0; e < 8; ++e) { bf0[e] = bl0[e]; bf0[8 + e] = bh0[e]; bf1[e] = bl1[e]; bf1[8 + e] = bh1[e]; }
        v8f acc0 = {}, acc1 = {};
        acc0 = WMMA_F16(a0[0], bf0, acc0); acc0 = WMMA_F16(a0[1], bf1, acc0);
        acc1 = WMMA_F16(a1[0], bf0, acc1); acc1 = WMMA_F16(a1[1], bf1, acc1);
        __syncthreads();
#pragma unroll
        for (int r = 0; r < 8; ++r) {
            stile[0][r + 8 * h][m] = acc0[r];
            stile[1][r + 8 * h][m] = acc1[r];
        }
        __syncthreads();
#pragma unroll 4
        for (int j = 0; j < 16; ++j) {
            float d = xr + sxx[col0 + j] - 2.0f * stile[myt][myr][j];
            if (d < worst) {
                bd[wi] = d; bi[wi] = col0 + j;
                worst = bd[0]; wi = 0;
#pragma unroll
                for (int u = 1; u < K_; ++u) {
                    if (bd[u] >= worst) { worst = bd[u]; wi = u; }
                }
            }
        }
    }

    int* op = idx + (size_t)(b * N_ + row0 + lane) * K_;
#pragma unroll
    for (int i = 0; i < K_; ++i) op[i] = bi[i];
}

// ---------------------------------------------------------------------------
// 5) BN statistics: out[b,n,k,o] = z[b,idx,o] + y[b,n,o], accumulate
//    per-channel sum / sumsq (never materializing out).
// ---------------------------------------------------------------------------
__global__ void __launch_bounds__(64) bn_stats(const float* __restrict__ z,
                                               const float* __restrict__ y,
                                               const int* __restrict__ idx,
                                               float* __restrict__ stats) {
    int o  = threadIdx.x;               // channel
    int p0 = blockIdx.x * 32;           // 32 points per block
    float s = 0.f, s2 = 0.f;
    for (int j = 0; j < 32; ++j) {
        int p = p0 + j;
        int b = p >> 12;
        float yv = y[(size_t)p * CO_ + o];
        const int* ip = idx + (size_t)p * K_;
#pragma unroll 4
        for (int kk = 0; kk < K_; ++kk) {
            int nb = ip[kk];
            float v = z[((size_t)(b * N_ + nb)) * CO_ + o] + yv;
            s += v; s2 += v * v;
        }
    }
    atomicAdd(&stats[o], s);
    atomicAdd(&stats[64 + o], s2);
}

__global__ void stats_fin(const float* __restrict__ stats,
                          const float* __restrict__ gamma,
                          const float* __restrict__ beta,
                          float* __restrict__ st) {
    int o = threadIdx.x;                // 64 threads
    float cnt  = (float)B_ * (float)N_ * (float)K_;
    float mean = stats[o] / cnt;
    float var  = stats[64 + o] / cnt - mean * mean;
    float s    = gamma[o] * rsqrtf(var + 1e-5f);
    st[o]      = s;
    st[64 + o] = beta[o] - mean * s;
}

// ---------------------------------------------------------------------------
// 6) Normalize + ReLU + max over k + transpose to (B, C_out, N).
//    val = z*s + (y*s + t); max-with-0 init fuses the ReLU.
// ---------------------------------------------------------------------------
__global__ void __launch_bounds__(256) finalize(const float* __restrict__ z,
                                                const float* __restrict__ y,
                                                const int* __restrict__ idx,
                                                const float* __restrict__ st,
                                                float* __restrict__ out) {
    __shared__ float ss[64], tt[64];
    int tid = threadIdx.x;
    if (tid < 64) { ss[tid] = st[tid]; tt[tid] = st[64 + tid]; }
    __syncthreads();

    int p = blockIdx.x * 256 + tid;     // b*N + n
    int b = p >> 12;
    int n = p & (N_ - 1);

    int ids[K_];
    const int* ip = idx + (size_t)p * K_;
#pragma unroll
    for (int kk = 0; kk < K_; ++kk) ids[kk] = b * N_ + ip[kk];

    const float* yrow = y + (size_t)p * CO_;
#pragma unroll
    for (int oc = 0; oc < 4; ++oc) {
        float pv[16], mv[16], sv[16];
#pragma unroll
        for (int e = 0; e < 16; ++e) {
            int o = oc * 16 + e;
            sv[e] = ss[o];
            pv[e] = yrow[o] * sv[e] + tt[o];
            mv[e] = 0.f;                         // ReLU floor
        }
        for (int kk = 0; kk < K_; ++kk) {
            const float4* zr = (const float4*)(z + (size_t)ids[kk] * CO_ + oc * 16);
#pragma unroll
            for (int q4 = 0; q4 < 4; ++q4) {
                float4 zv = zr[q4];
                int e = q4 * 4;
                mv[e + 0] = fmaxf(mv[e + 0], zv.x * sv[e + 0] + pv[e + 0]);
                mv[e + 1] = fmaxf(mv[e + 1], zv.y * sv[e + 1] + pv[e + 1]);
                mv[e + 2] = fmaxf(mv[e + 2], zv.z * sv[e + 2] + pv[e + 2]);
                mv[e + 3] = fmaxf(mv[e + 3], zv.w * sv[e + 3] + pv[e + 3]);
            }
        }
#pragma unroll
        for (int e = 0; e < 16; ++e) {
            int o = oc * 16 + e;
            out[(size_t)b * CO_ * N_ + (size_t)o * N_ + n] = mv[e];
        }
    }
}

// ---------------------------------------------------------------------------
extern "C" void kernel_launch(void* const* d_in, const int* in_sizes, int n_in,
                              void* d_out, int out_size, void* d_ws, size_t ws_size,
                              hipStream_t stream) {
    const float* x     = (const float*)d_in[0];
    const float* W     = (const float*)d_in[1];
    const float* gamma = (const float*)d_in[2];
    const float* beta  = (const float*)d_in[3];
    float* out = (float*)d_out;

    char* ws = (char*)d_ws;
    size_t off = 0;
    auto alloc = [&](size_t bytes) -> void* {
        void* p = ws + off;
        off += (bytes + 255) & ~(size_t)255;
        return p;
    };

    _Float16* xh  = (_Float16*)alloc((size_t)B_ * N_ * C_ * sizeof(_Float16));
    float*    xx  = (float*)   alloc((size_t)B_ * N_ * sizeof(float));
    _Float16* w1f = (_Float16*)alloc((size_t)256 * 16 * sizeof(_Float16));
    _Float16* wdf = (_Float16*)alloc((size_t)256 * 16 * sizeof(_Float16));
    float*    z   = (float*)   alloc((size_t)B_ * N_ * CO_ * sizeof(float));
    float*    y   = (float*)   alloc((size_t)B_ * N_ * CO_ * sizeof(float));
    int*      idx = (int*)     alloc((size_t)B_ * N_ * K_ * sizeof(int));
    float*  stats = (float*)   alloc(128 * sizeof(float));
    float*     st = (float*)   alloc(128 * sizeof(float));

    prep_x   <<<B_ * N_ / 256, 256, 0, stream>>>(x, xh, xx);
    prep_w   <<<1, 256, 0, stream>>>(W, w1f, wdf);
    zy_gemm  <<<B_ * N_ / 16, 32, 0, stream>>>(xh, w1f, wdf, z, y);
    knn_kernel<<<B_ * (N_ / 32), 32, 0, stream>>>(xh, xx, idx);
    hipMemsetAsync(stats, 0, 128 * sizeof(float), stream);
    bn_stats <<<B_ * N_ / 32, 64, 0, stream>>>(z, y, idx, stats);
    stats_fin<<<1, 64, 0, stream>>>(stats, gamma, beta, st);
    finalize <<<B_ * N_ / 256, 256, 0, stream>>>(z, y, idx, st, out);
}
